// Encoder_19705309954220
// MI455X (gfx1250) — matrix operand order repaired and checked by reading, not compiled, not run on hardware
//
#include <hip/hip_runtime.h>

// LSTM: B=131072, T=7, F=60, UNITS=256. Gate order i,f,g,o in 4*UNITS axis.
// bf16 WMMA. 32 waves/WG in 16 pairs: even wave of a pair computes gates i,f
// for 16 hidden units, odd wave computes g,o. Per wave: W frags (32 VGPRs) +
// U ktiles 0-3 (64 VGPRs) register-resident; U ktiles 4-7 in LDS (256 KB).
// Gates recombined through a small LDS exchange; h double-buffered in LDS;
// c stays in f32 C-layout registers. Persistent grid, weights loaded once/WG.
// Activations use v_exp_f32 + v_rcp_f32 (no IEEE-division expansion).

#define B_TOTAL   131072
#define T_STEPS   7
#define F_IN      60
#define UNITS     256
#define KX        64                      // F padded to multiple of 32
#define BLK_ROWS  16
#define NBLK      (B_TOTAL / BLK_ROWS)    // 8192
#define NWAVES    32
#define THREADS   (NWAVES * 32)           // 1024 threads = 32 wave32
#define GRID      1024

// dynamic LDS layout (ushort units unless noted)
#define U_LDS_ELEMS   (1024 * 128)                // K=128..255 of U: 131072 (256 KB)
#define X_LDS_ELEMS   (T_STEPS * BLK_ROWS * KX)   // 7168 (14 KB)
#define H_LDS_ELEMS   (BLK_ROWS * UNITS)          // 4096 (8 KB each, x2)
#define XCHG_OFF_US   (U_LDS_ELEMS + X_LDS_ELEMS + 2 * H_LDS_ELEMS) // 146432
#define XCHG_FLOATS   (16 * 2 * 256)              // 16 pairs x 2 tiles x 256 = 32 KB
#define SMEM_BYTES    (XCHG_OFF_US * 2 + XCHG_FLOATS * 4)  // 325632 B <= 320 KB

typedef __attribute__((ext_vector_type(16))) __bf16 v16bf;
typedef __attribute__((ext_vector_type(8)))  float  v8f;

union Frag {
  v16bf v;
  uint4 q[2];
};

__device__ __forceinline__ unsigned short f2bf(float f) {
  unsigned u = __float_as_uint(f);
  u += 0x7FFFu + ((u >> 16) & 1u);        // round to nearest even
  return (unsigned short)(u >> 16);
}

// v_rcp_f32: ~1 ulp, plenty for LSTM gates (inputs are bf16-rounded anyway).
__device__ __forceinline__ float fast_rcp(float x) {
  return __builtin_amdgcn_rcpf(x);
}
__device__ __forceinline__ float sigm(float z) {
  return fast_rcp(1.0f + __expf(-z));
}
__device__ __forceinline__ float tanh_fast(float z) {
  return __builtin_fmaf(2.0f, fast_rcp(1.0f + __expf(-2.0f * z)), -1.0f);
}

// Pre-transpose weights to bf16, K-contiguous per output column:
// Wt[n][k] (n<1024, k<64, zero-padded past F_IN), Ut[n][k] (n<1024, k<256).
__global__ __launch_bounds__(256) void convert_weights(
    const float* __restrict__ W, const float* __restrict__ U,
    unsigned short* __restrict__ Wt, unsigned short* __restrict__ Ut)
{
  const int i = blockIdx.x * blockDim.x + threadIdx.x;
  if (i < 1024 * KX) {
    const int n = i >> 6, k = i & 63;
    Wt[i] = (k < F_IN) ? f2bf(W[k * 1024 + n]) : (unsigned short)0;
  } else if (i < 1024 * KX + 1024 * UNITS) {
    const int j = i - 1024 * KX;
    const int n = j >> 8, k = j & 255;
    Ut[j] = f2bf(U[k * 1024 + n]);
  }
}

__global__ __launch_bounds__(THREADS, 1) void lstm_fused(
    const float* __restrict__ x, const float* __restrict__ b,
    const unsigned short* __restrict__ Wt, const unsigned short* __restrict__ Ut,
    float* __restrict__ out)
{
  extern __shared__ __align__(16) unsigned short smem[];
  unsigned short* ulds  = smem;                       // [1024 cols][128] (K hi half)
  unsigned short* xlds  = smem + U_LDS_ELEMS;         // [7][16][64]
  unsigned short* hlds0 = xlds + X_LDS_ELEMS;         // [16][256] x2
  unsigned short* hlds1 = hlds0 + H_LDS_ELEMS;
  float* xchg = (float*)(smem + XCHG_OFF_US);         // [16 pairs][2][256]

  const int tid  = threadIdx.x;
  const int lane = tid & 31;               // wave32
  const int wave = tid >> 5;
  const int pair = wave >> 1;
  const int half = wave & 1;               // 0: gates i,f   1: gates g,o
  const int up16 = pair * 16;              // this pair's hidden-unit base
  const int l15  = lane & 15;
  const int lhi  = lane >> 4;
  const int kb16 = lhi * 16;               // B-frag K base (lane half)
  const int kb8  = lhi * 16;               // A-frag K base (lane half)
  const int gbase = half * 2;              // first gate handled by this wave

  // Bias per lane for this wave's two gate columns.
  float bv[2];
#pragma unroll
  for (int gg = 0; gg < 2; ++gg)
    bv[gg] = b[(gbase + gg) * UNITS + up16 + l15];

  // Stage K=128..255 half of U (256 KB) into LDS: 16384 uint4, 16/thread.
  for (int e = tid; e < U_LDS_ELEMS / 8; e += THREADS) {
    const int col = e >> 4;
    const int rem = e & 15;
    ((uint4*)ulds)[e] = *(const uint4*)(Ut + col * UNITS + 128 + rem * 8);
  }

  // Register-resident fragments (B layout: per lane, column n = n0+(lane&15),
  // 16 consecutive K packed 2/VGPR): W 2kt x 2g, U ktiles 0-3 x 2g.
  Frag wfrag[2][2];
  Frag ufrag[4][2];
#pragma unroll
  for (int gg = 0; gg < 2; ++gg) {
    const int col = (gbase + gg) * UNITS + up16 + l15;
    const unsigned short* wp = Wt + col * KX;
    const unsigned short* up = Ut + col * UNITS;
#pragma unroll
    for (int kt = 0; kt < 2; ++kt) {
      const unsigned short* p = wp + kt * 32 + kb16;
      wfrag[kt][gg].q[0] = *(const uint4*)(p);
      wfrag[kt][gg].q[1] = *(const uint4*)(p + 8);
    }
#pragma unroll
    for (int kt = 0; kt < 4; ++kt) {
      const unsigned short* p = up + kt * 32 + kb16;
      ufrag[kt][gg].q[0] = *(const uint4*)(p);
      ufrag[kt][gg].q[1] = *(const uint4*)(p + 8);
    }
  }

  // LDS base for this wave's U hi-half fragment columns (per gate).
  const unsigned short* ucol[2];
#pragma unroll
  for (int gg = 0; gg < 2; ++gg)
    ucol[gg] = ulds + ((gbase + gg) * UNITS + up16 + l15) * 128 + kb16;

  float* xq = xchg + pair * 512;           // [2][256] exchange for this pair

  // Persistent loop over 16-row batch blocks.
  for (int blk = blockIdx.x; blk < NBLK; blk += gridDim.x) {
    const int row0 = blk * BLK_ROWS;
    __syncthreads();   // U staged / previous block done with LDS
    // Stage x tile into LDS as bf16 (K padded to 64): 7168 = 7*1024 exact.
    for (int e = tid; e < X_LDS_ELEMS; e += THREADS) {
      const int t = e >> 10;
      const int m = (e >> 6) & 15;
      const int k = e & 63;
      float v = 0.0f;
      if (k < F_IN) v = x[(long)(row0 + m) * (T_STEPS * F_IN) + t * F_IN + k];
      xlds[e] = f2bf(v);
    }
    __syncthreads();

    v8f creg, hreg;                        // valid in even waves only
#pragma unroll
    for (int e = 0; e < 8; ++e) { creg[e] = 0.0f; hreg[e] = 0.0f; }

    for (int t = 0; t < T_STEPS; ++t) {
      v8f acc[2];
#pragma unroll
      for (int gg = 0; gg < 2; ++gg) {
#pragma unroll
        for (int e = 0; e < 8; ++e) acc[gg][e] = bv[gg];
      }

      // z += x_t @ W
#pragma unroll
      for (int kt = 0; kt < 2; ++kt) {
        Frag fa;
        const unsigned short* p = &xlds[(t * BLK_ROWS + l15) * KX + kt * 32 + kb8];
        fa.q[0] = *(const uint4*)(p);
        fa.q[1] = *(const uint4*)(p + 8);
#pragma unroll
        for (int gg = 0; gg < 2; ++gg)
          acc[gg] = __builtin_amdgcn_wmma_f32_16x16x32_bf16(
              false, fa.v, false, wfrag[kt][gg].v, (short)0, acc[gg], false, false);
      }

      // z += h_{t-1} @ U
      if (t > 0) {
        const unsigned short* hb = (t & 1) ? hlds0 : hlds1;  // buf[(t-1)&1]
        // ktiles 0-3: register-resident U fragments
#pragma unroll
        for (int kt = 0; kt < 4; ++kt) {
          Frag fa;
          const unsigned short* p = hb + l15 * UNITS + kt * 32 + kb8;
          fa.q[0] = *(const uint4*)(p);
          fa.q[1] = *(const uint4*)(p + 8);
#pragma unroll
          for (int gg = 0; gg < 2; ++gg)
            acc[gg] = __builtin_amdgcn_wmma_f32_16x16x32_bf16(
                false, fa.v, false, ufrag[kt][gg].v, (short)0, acc[gg], false, false);
        }
        // ktiles 4-7: U fragments streamed from LDS
#pragma unroll
        for (int kt = 4; kt < 8; ++kt) {
          Frag fa;
          const unsigned short* p = hb + l15 * UNITS + kt * 32 + kb8;
          fa.q[0] = *(const uint4*)(p);
          fa.q[1] = *(const uint4*)(p + 8);
#pragma unroll
          for (int gg = 0; gg < 2; ++gg) {
            Frag fu;
            const unsigned short* q = ucol[gg] + (kt - 4) * 32;
            fu.q[0] = *(const uint4*)(q);
            fu.q[1] = *(const uint4*)(q + 8);
            acc[gg] = __builtin_amdgcn_wmma_f32_16x16x32_bf16(
                false, fa.v, false, fu.v, (short)0, acc[gg], false, false);
          }
        }
      }

      // Gate exchange: odd wave activates g,o and publishes; even wave
      // combines with i,f, updates c,h. (lane,e) -> (m=e+8*lhi, n=up16+l15)
      // identically in both waves of a pair.
      if (half) {
#pragma unroll
        for (int e = 0; e < 8; ++e) {
          xq[e * 32 + lane]       = tanh_fast(acc[0][e]);   // g
          xq[256 + e * 32 + lane] = sigm(acc[1][e]);        // o
        }
      }
      __syncthreads();
      unsigned short* hw = (t & 1) ? hlds1 : hlds0;          // buf[t&1]
      if (!half) {
#pragma unroll
        for (int e = 0; e < 8; ++e) {
          const float iv = sigm(acc[0][e]);
          const float fv = sigm(acc[1][e]);
          const float gv = xq[e * 32 + lane];
          const float ov = xq[256 + e * 32 + lane];
          const float cn = fv * creg[e] + iv * gv;
          creg[e] = cn;
          const float hn = ov * tanh_fast(cn);
          hreg[e] = hn;
          hw[(e + lhi * 8) * UNITS + up16 + l15] = f2bf(hn);
        }
      }
      __syncthreads();                     // h_t visible; xchg reusable
    }

    // Outputs: d_out = [h | h | c], row-major [B, 256] fp32 (even waves).
    if (!half) {
      const int nglob = up16 + l15;
#pragma unroll
      for (int e = 0; e < 8; ++e) {
        const long r = (long)row0 + lhi * 8 + e;
        const long o = r * UNITS + nglob;
        const float hv = hreg[e];
        out[o] = hv;
        out[(long)B_TOTAL * UNITS + o] = hv;
        out[2L * (long)B_TOTAL * UNITS + o] = creg[e];
      }
    }
  }
}

extern "C" void kernel_launch(void* const* d_in, const int* in_sizes, int n_in,
                              void* d_out, int out_size, void* d_ws, size_t ws_size,
                              hipStream_t stream) {
  const float* x = (const float*)d_in[0];   // [131072, 7, 60]
  const float* W = (const float*)d_in[1];   // [60, 1024]
  const float* U = (const float*)d_in[2];   // [256, 1024]
  const float* b = (const float*)d_in[3];   // [1024]
  float* out = (float*)d_out;               // [3 * 131072 * 256]

  unsigned short* Wt = (unsigned short*)d_ws;            // 1024*64  bf16 = 128 KB
  unsigned short* Ut = Wt + 1024 * KX;                   // 1024*256 bf16 = 512 KB

  // 318 KB of the 320 KB WGP LDS.
  (void)hipFuncSetAttribute((const void*)lstm_fused,
                            hipFuncAttributeMaxDynamicSharedMemorySize,
                            SMEM_BYTES);

  const int conv_total = 1024 * KX + 1024 * UNITS;       // 327680
  convert_weights<<<(conv_total + 255) / 256, 256, 0, stream>>>(W, U, Wt, Ut);
  lstm_fused<<<GRID, THREADS, SMEM_BYTES, stream>>>(x, b, Wt, Ut, out);
}
